// HAN_66881230733580
// MI455X (gfx1250) — compile-verified, matrix operand order checked by reading, not run on hardware
//
#include <hip/hip_runtime.h>
#include <hip/hip_bf16.h>
#include <math.h>

#define NN   50000
#define EE   400000
#define PP   3
#define HH   4
#define DD   32
#define INF_ 128   // in_size
#define HD_  128   // H*D

typedef __attribute__((ext_vector_type(16))) _Float16 v16h;
typedef __attribute__((ext_vector_type(8)))  _Float16 v8h;
typedef __attribute__((ext_vector_type(8)))  float    v8f;

// ---------------- helpers ----------------
__device__ __forceinline__ unsigned enc_f32(float f) {
    unsigned u = __float_as_uint(f);
    return (u & 0x80000000u) ? ~u : (u | 0x80000000u);
}
__device__ __forceinline__ float dec_f32(unsigned k) {
    unsigned u = (k & 0x80000000u) ? (k & 0x7fffffffu) : ~k;
    return __uint_as_float(u);
}

// Load a v16h A/B fragment for v_wmma_f32_16x16x32_f16.
// 16-bit A layout (ISA 7.12.2): lanes 0-15 hold K={0..7,16..23}, lanes 16-31
// hold K={8..15,24..31}; row/col = lane%16. Two contiguous 16B loads.
__device__ __forceinline__ v16h load_frag(const _Float16* rowptr, int kbase, int off) {
    v8h lo = *(const v8h*)(rowptr + kbase + off);
    v8h hi = *(const v8h*)(rowptr + kbase + off + 16);
    v16h r;
#pragma unroll
    for (int j = 0; j < 8; ++j) { r[j] = lo[j]; r[8 + j] = hi[j]; }
    return r;
}

// ---------------- kernel 0: weight prep (transpose + f16) ----------------
__global__ __launch_bounds__(256) void prep_weights(const float* __restrict__ gat_W,
                                                    const float* __restrict__ sa_w1,
                                                    _Float16* __restrict__ WtH,
                                                    _Float16* __restrict__ saW1t) {
    int t = blockIdx.x * 256 + threadIdx.x;
    if (t < PP * 128 * 128) {
        int p = t >> 14, r = t & 16383, n = r >> 7, k = r & 127;
        WtH[t] = (_Float16)gat_W[p * 16384 + k * 128 + n]; // WtH[p][n][k] = W[p][k][n]
    }
    if (t < 128 * 128) {
        int n = t >> 7, k = t & 127;
        saW1t[t] = (_Float16)sa_w1[k * 128 + n];
    }
}

// ---------------- kernel 1: feature attention + concat -> x (f16) ----------------
__global__ __launch_bounds__(256) void feat_attn(const float* __restrict__ h,
                                                 const float* __restrict__ cin,
                                                 const float* __restrict__ w1,
                                                 const float* __restrict__ b1,
                                                 const float* __restrict__ w2,
                                                 _Float16* __restrict__ xh) {
    __shared__ float w1s[672], b1s[16], w2s[16];
    __shared__ float fbuf[8][20 * 42];
    int tid = threadIdx.x;
    for (int i = tid; i < 672; i += 256) w1s[i] = w1[i];
    if (tid < 16) { b1s[tid] = b1[tid]; w2s[tid] = w2[tid]; }
    int wv = tid >> 5, lane = tid & 31;
    int n = blockIdx.x * 8 + wv;            // 6250 blocks * 8 waves = 50000 exactly
    float* fb = fbuf[wv];
    const float* base = cin + (size_t)n * 840;
    for (int i = lane; i < 840; i += 32) fb[i] = base[i];
    __syncthreads();

    float wlogit = -3.0e38f;
    if (lane < 20) {
        float acc = 0.f;
#pragma unroll
        for (int j = 0; j < 16; ++j) {
            float s = b1s[j];
            for (int k = 0; k < 42; ++k) s += fb[lane * 42 + k] * w1s[k * 16 + j];
            acc += tanhf(s) * w2s[j];
        }
        wlogit = acc;
    }
    float m = wlogit;
    for (int o = 16; o; o >>= 1) m = fmaxf(m, __shfl_xor(m, o, 32));
    float ee = (lane < 20) ? expf(wlogit - m) : 0.f;
    float ssum = ee;
    for (int o = 16; o; o >>= 1) ssum += __shfl_xor(ssum, o, 32);
    float beta = ee / ssum;

    float c0 = 0.f, c1 = 0.f;
    for (int i = 0; i < 20; ++i) {
        float bi = __shfl(beta, i, 32);
        c0 += bi * fb[i * 42 + lane];
        if (lane < 10) c1 += bi * fb[i * 42 + lane + 32];
    }
    _Float16* xr = xh + (size_t)n * 128;
    for (int j = lane; j < 86; j += 32) xr[j] = (_Float16)h[(size_t)n * 86 + j];
    xr[86 + lane] = (_Float16)c0;
    if (lane < 10) xr[118 + lane] = (_Float16)c1;
}

// ---------------- kernel 2: WMMA GEMM  feat[p] = x @ W_p ----------------
__global__ __launch_bounds__(256) void gat_gemm(const _Float16* __restrict__ xh,
                                                const _Float16* __restrict__ WtH,
                                                float* __restrict__ feat) {
    int wv = threadIdx.x >> 5, lane = threadIdx.x & 31;
    int task = blockIdx.x * 8 + wv;
    if (task >= PP * (NN / 16)) return;
    int p = task / (NN / 16), rt = task % (NN / 16);
    int row0 = rt * 16;
    int mrow = lane & 15;
    int off = (lane < 16) ? 0 : 8;
    const _Float16* arow = xh + (size_t)(row0 + mrow) * 128;
    const _Float16* bbase = WtH + (size_t)p * 16384;

    v8f acc[8] = {};
#pragma unroll
    for (int kt = 0; kt < 4; ++kt) {
        int kbase = kt * 32;
        v16h a = load_frag(arow, kbase, off);
#pragma unroll
        for (int nt = 0; nt < 8; ++nt) {
            const _Float16* brow = bbase + (size_t)(nt * 16 + mrow) * 128;
            v16h b = load_frag(brow, kbase, off);
            acc[nt] = __builtin_amdgcn_wmma_f32_16x16x32_f16(
                false, a, false, b, (short)0, acc[nt], false, false);
        }
    }
    float* fout = feat + ((size_t)p * NN + row0) * 128;
    int rbase = (lane < 16) ? 0 : 8;
#pragma unroll
    for (int nt = 0; nt < 8; ++nt)
#pragma unroll
        for (int i = 0; i < 8; ++i)
            fout[(size_t)(rbase + i) * 128 + nt * 16 + mrow] = acc[nt][i];
}

// ---------------- kernel 3: el/er dot products ----------------
__global__ __launch_bounds__(256) void compute_eler(const float* __restrict__ feat,
                                                    const float* __restrict__ attn_l,
                                                    const float* __restrict__ attn_r,
                                                    float* __restrict__ el,
                                                    float* __restrict__ er) {
    int wv = threadIdx.x >> 5, lane = threadIdx.x & 31;
    int t = blockIdx.x * 8 + wv;
    if (t >= PP * NN) return;
    int p = t / NN;
    const float* fr = feat + (size_t)t * 128;
    const float* al = attn_l + p * 128;
    const float* ar = attn_r + p * 128;
    float elp[4], erp[4];
#pragma unroll
    for (int j = 0; j < 4; ++j) {
        float v = fr[j * 32 + lane];
        elp[j] = v * al[j * 32 + lane];
        erp[j] = v * ar[j * 32 + lane];
    }
#pragma unroll
    for (int j = 0; j < 4; ++j)
        for (int o = 16; o; o >>= 1) {
            elp[j] += __shfl_xor(elp[j], o, 32);
            erp[j] += __shfl_xor(erp[j], o, 32);
        }
    if (lane < 4) el[(size_t)t * 4 + lane] = elp[lane];
    else if (lane < 8) er[(size_t)t * 4 + (lane - 4)] = erp[lane - 4];
}

// ---------------- fills ----------------
__global__ void fill_u32(unsigned* __restrict__ p, unsigned v, long long count) {
    long long i = (long long)blockIdx.x * blockDim.x + threadIdx.x;
    if (i < count) p[i] = v;
}

// ---------------- kernel 4: edge pass A (segment max via atomic u32 max) ----------------
__global__ __launch_bounds__(256) void edge_max(const int* __restrict__ src, const int* __restrict__ dst,
                                                const float* __restrict__ el, const float* __restrict__ er,
                                                unsigned* __restrict__ mEnc) {
    int t = blockIdx.x * 256 + threadIdx.x;
    if (t >= PP * EE) return;
    int p = t / EE;
    int s = src[t], d = dst[t];
    const float* elr = el + ((size_t)p * NN + s) * 4;
    const float* err = er + ((size_t)p * NN + d) * 4;
    unsigned* mr = mEnc + ((size_t)p * NN + d) * 4;
#pragma unroll
    for (int j = 0; j < 4; ++j) {
        float v = elr[j] + err[j];
        v = v > 0.f ? v : 0.2f * v;
        atomicMax(&mr[j], enc_f32(v));
    }
}

// ---------------- kernel 5: edge pass B (exp + segment sum) ----------------
__global__ __launch_bounds__(256) void edge_exp(const int* __restrict__ src, const int* __restrict__ dst,
                                                const float* __restrict__ el, const float* __restrict__ er,
                                                const unsigned* __restrict__ mEnc,
                                                float* __restrict__ zsum, float* __restrict__ eeBuf) {
    int t = blockIdx.x * 256 + threadIdx.x;
    if (t >= PP * EE) return;
    int p = t / EE;
    int s = src[t], d = dst[t];
    const float* elr = el + ((size_t)p * NN + s) * 4;
    const float* err = er + ((size_t)p * NN + d) * 4;
    const unsigned* mr = mEnc + ((size_t)p * NN + d) * 4;
    float* zr = zsum + ((size_t)p * NN + d) * 4;
#pragma unroll
    for (int j = 0; j < 4; ++j) {
        float v = elr[j] + err[j];
        v = v > 0.f ? v : 0.2f * v;
        float e2 = expf(v - dec_f32(mr[j]));
        eeBuf[(size_t)t * 4 + j] = e2;
        unsafeAtomicAdd(&zr[j], e2);
    }
}

// ---------------- kernel 6: edge pass C (alpha * feat[src] scatter-add) ----------------
__global__ __launch_bounds__(256) void edge_scatter(const int* __restrict__ src, const int* __restrict__ dst,
                                                    const float* __restrict__ feat,
                                                    const float* __restrict__ zsum,
                                                    const float* __restrict__ eeBuf,
                                                    float* __restrict__ outacc) {
    int wv = threadIdx.x >> 5, lane = threadIdx.x & 31;
    long long t = (long long)blockIdx.x * 8 + wv;
    if (t >= (long long)PP * EE) return;
    int p = (int)(t / EE);
    int s = src[t], d = dst[t];
    const float* fr = feat + ((size_t)p * NN + s) * 128;
    float* outr = outacc + ((size_t)p * NN + d) * 128;
    const float* zr = zsum + ((size_t)p * NN + d) * 4;
    const float* eer = eeBuf + (size_t)t * 4;
#pragma unroll
    for (int j = 0; j < 4; ++j) {
        float alpha = eer[j] / zr[j];
        int c = j * 32 + lane;
        unsafeAtomicAdd(&outr[c], alpha * fr[c]);
    }
}

// ---------------- kernel 7: ELU in place + f16 copy for semantic GEMM ----------------
__global__ __launch_bounds__(256) void elu_convert(float* __restrict__ outacc,
                                                   _Float16* __restrict__ zh, long long count) {
    long long i = (long long)blockIdx.x * 256 + threadIdx.x;
    if (i < count) {
        float v = outacc[i];
        v = v > 0.f ? v : expm1f(v);
        outacc[i] = v;
        zh[i] = (_Float16)v;
    }
}

// ---------------- kernel 8: semantic attention GEMM + global reduce ----------------
__global__ __launch_bounds__(256) void sem_gemm(const _Float16* __restrict__ zh,
                                                const _Float16* __restrict__ saW1t,
                                                const float* __restrict__ sa_b1,
                                                const float* __restrict__ sa_w2,
                                                float* __restrict__ wsum) {
    int wv = threadIdx.x >> 5, lane = threadIdx.x & 31;
    int task = blockIdx.x * 8 + wv;
    if (task >= PP * (NN / 16)) return;
    int p = task / (NN / 16), rt = task % (NN / 16);
    int row0 = rt * 16;
    int mrow = lane & 15;
    int off = (lane < 16) ? 0 : 8;
    const _Float16* arow = zh + ((size_t)p * NN + row0 + mrow) * 128;

    v8f acc[8] = {};
#pragma unroll
    for (int kt = 0; kt < 4; ++kt) {
        int kbase = kt * 32;
        v16h a = load_frag(arow, kbase, off);
#pragma unroll
        for (int nt = 0; nt < 8; ++nt) {
            const _Float16* brow = saW1t + (size_t)(nt * 16 + mrow) * 128;
            v16h b = load_frag(brow, kbase, off);
            acc[nt] = __builtin_amdgcn_wmma_f32_16x16x32_f16(
                false, a, false, b, (short)0, acc[nt], false, false);
        }
    }
    float part = 0.f;
#pragma unroll
    for (int nt = 0; nt < 8; ++nt) {
        int col = nt * 16 + mrow;
        float b1c = sa_b1[col], w2c = sa_w2[col];
#pragma unroll
        for (int i = 0; i < 8; ++i) part += tanhf(acc[nt][i] + b1c) * w2c;
    }
    for (int o = 16; o; o >>= 1) part += __shfl_xor(part, o, 32);
    if (lane == 0) unsafeAtomicAdd(&wsum[p], part);
}

// ---------------- kernel 9: softmax over 3 semantic logits ----------------
__global__ void softmax3(const float* __restrict__ wsum, float* __restrict__ bsem) {
    if (threadIdx.x == 0) {
        float a = wsum[0] / (float)NN, b = wsum[1] / (float)NN, c = wsum[2] / (float)NN;
        float m = fmaxf(a, fmaxf(b, c));
        float ea = expf(a - m), eb = expf(b - m), ec = expf(c - m);
        float s = ea + eb + ec;
        bsem[0] = ea / s; bsem[1] = eb / s; bsem[2] = ec / s;
    }
}

// ---------------- kernel 10: final weighted mix ----------------
__global__ __launch_bounds__(256) void final_mix(const float* __restrict__ z,
                                                 const float* __restrict__ bsem,
                                                 float* __restrict__ out) {
    long long i = (long long)blockIdx.x * 256 + threadIdx.x;
    if (i < (long long)NN * 128) {
        out[i] = bsem[0] * z[i]
               + bsem[1] * z[i + (long long)NN * 128]
               + bsem[2] * z[i + 2LL * NN * 128];
    }
}

// ---------------- launcher ----------------
extern "C" void kernel_launch(void* const* d_in, const int* in_sizes, int n_in,
                              void* d_out, int out_size, void* d_ws, size_t ws_size,
                              hipStream_t stream) {
    (void)in_sizes; (void)n_in; (void)out_size; (void)ws_size;
    const float* h       = (const float*)d_in[0];
    const float* cin     = (const float*)d_in[1];
    const float* fa_w1   = (const float*)d_in[2];
    const float* fa_b1   = (const float*)d_in[3];
    const float* fa_w2   = (const float*)d_in[4];
    const float* gat_W   = (const float*)d_in[5];
    const float* attn_l  = (const float*)d_in[6];
    const float* attn_r  = (const float*)d_in[7];
    const float* sa_w1   = (const float*)d_in[8];
    const float* sa_b1   = (const float*)d_in[9];
    const float* sa_w2   = (const float*)d_in[10];
    const int*   src     = (const int*)d_in[11];
    const int*   dst     = (const int*)d_in[12];
    float* out = (float*)d_out;

    char* w = (char*)d_ws;
    auto carve = [&](size_t bytes) {
        char* p = w;
        w += (bytes + 255) & ~(size_t)255;
        return p;
    };
    _Float16* xh     = (_Float16*)carve((size_t)NN * 128 * 2);
    _Float16* WtH    = (_Float16*)carve((size_t)PP * 128 * 128 * 2);
    _Float16* saW1t  = (_Float16*)carve((size_t)128 * 128 * 2);
    float*    feat   = (float*)   carve((size_t)PP * NN * 128 * 4);
    float*    el     = (float*)   carve((size_t)PP * NN * 4 * 4);
    float*    er     = (float*)   carve((size_t)PP * NN * 4 * 4);
    unsigned* mEnc   = (unsigned*)carve((size_t)PP * NN * 4 * 4);
    float*    zsum   = (float*)   carve((size_t)PP * NN * 4 * 4);
    float*    eeBuf  = (float*)   carve((size_t)PP * EE * 4 * 4);
    float*    outacc = (float*)   carve((size_t)PP * NN * 128 * 4);
    float*    wsumb  = (float*)   carve(256);   // [0..2]=wsum, [4..6]=bsem
    _Float16* zh     = (_Float16*)feat;         // reuse feat region after edge passes

    // 0. weight prep
    prep_weights<<<(PP * 16384 + 255) / 256, 256, 0, stream>>>(gat_W, sa_w1, WtH, saW1t);
    // 1. feature attention -> x (f16)
    feat_attn<<<NN / 8, 256, 0, stream>>>(h, cin, fa_w1, fa_b1, fa_w2, xh);
    // init accumulators
    fill_u32<<<(PP * NN * 4 + 255) / 256, 256, 0, stream>>>(mEnc, 0x007FFFFFu, (long long)PP * NN * 4); // enc(-inf)
    fill_u32<<<(PP * NN * 4 + 255) / 256, 256, 0, stream>>>((unsigned*)zsum, 0u, (long long)PP * NN * 4);
    fill_u32<<<((long long)PP * NN * 128 + 255) / 256, 256, 0, stream>>>((unsigned*)outacc, 0u, (long long)PP * NN * 128);
    fill_u32<<<1, 64, 0, stream>>>((unsigned*)wsumb, 0u, 64);
    // 2. WMMA GEMM: feat = x @ W_p
    gat_gemm<<<(PP * (NN / 16) + 7) / 8, 256, 0, stream>>>(xh, WtH, feat);
    // 3. el / er
    compute_eler<<<(PP * NN + 7) / 8, 256, 0, stream>>>(feat, attn_l, attn_r, el, er);
    // 4-6. edge softmax + scatter
    edge_max<<<(PP * EE + 255) / 256, 256, 0, stream>>>(src, dst, el, er, mEnc);
    edge_exp<<<(PP * EE + 255) / 256, 256, 0, stream>>>(src, dst, el, er, mEnc, zsum, eeBuf);
    edge_scatter<<<(PP * EE + 7) / 8, 256, 0, stream>>>(src, dst, feat, zsum, eeBuf, outacc);
    // 7. ELU + f16 convert (zh aliases feat; feat no longer needed)
    elu_convert<<<((long long)PP * NN * 128 + 255) / 256, 256, 0, stream>>>(outacc, zh, (long long)PP * NN * 128);
    // 8. semantic attention WMMA GEMM + reduce
    sem_gemm<<<(PP * (NN / 16) + 7) / 8, 256, 0, stream>>>(zh, saW1t, sa_b1, sa_w2, wsumb);
    // 9. softmax over paths
    softmax3<<<1, 32, 0, stream>>>(wsumb, wsumb + 4);
    // 10. final mix
    final_mix<<<((long long)NN * 128 + 255) / 256, 256, 0, stream>>>(outacc, wsumb + 4, out);
}
